// TriDirectionalMamba_74887049773361
// MI455X (gfx1250) — matrix operand order repaired and checked by reading, not compiled, not run on hardware
//
#include <hip/hip_runtime.h>
#include <math.h>

// ---------------- types for WMMA ----------------
typedef __attribute__((ext_vector_type(16))) _Float16 v16h;
typedef __attribute__((ext_vector_type(8)))  _Float16 v8h;
typedef __attribute__((ext_vector_type(8)))  float    v8f;

// ---------------- problem constants ----------------
#define CC      64
#define NST     16
#define DIx     128
#define KCONV   4
#define DTRANK  4
#define LSEQ    48
#define DIMS    48
#define HWW     (48*48)
#define DHW     (48*48*48)
#define SEQS    (48*48)     // 2304 sequences / direction
#define TTOK    (SEQS*LSEQ) // 110592 tokens / direction

// index into x/out (B=1, C, D, H, W) for direction-dependent (seq s, pos l, chan c)
__device__ __forceinline__ int gidx(int dir, int s, int l, int c) {
  if (dir == 0) return c * DHW + l * HWW + s;
  if (dir == 1) return c * DHW + (s / DIMS) * HWW + l * DIMS + (s % DIMS);
  return c * DHW + s * DIMS + l;
}

// ---- CDNA5 async global->LDS copy (ASYNCcnt path, ISA §10 / 08_async_tensor) ----
__device__ __forceinline__ void async_b128(unsigned lds_off, const void* gaddr) {
  asm volatile("global_load_async_to_lds_b128 %0, %1, off"
               :: "v"(lds_off), "v"(gaddr) : "memory");
}
__device__ __forceinline__ void wait_async0() {
  asm volatile("s_wait_asynccnt 0x0" ::: "memory");
}
// cooperative copy of nchunks*16 bytes, 256 threads
__device__ __forceinline__ void coop_async_copy(const void* g, unsigned lds_base, int nchunks) {
  for (int i = threadIdx.x; i < nchunks; i += 256)
    async_b128(lds_base + i * 16, (const char*)g + i * 16);
}
__device__ __forceinline__ unsigned lds_off_of(const void* p) {
  return (unsigned)(unsigned long long)p;   // low 32 bits of flat addr = LDS byte offset
}

// ---------------- weight transpose / f16 convert ----------------
__global__ void k_prep(int dir, const float* __restrict__ in_proj,
                       const float* __restrict__ x_proj,
                       const float* __restrict__ out_proj,
                       _Float16* __restrict__ w1, _Float16* __restrict__ w2,
                       _Float16* __restrict__ w3) {
  int i = blockIdx.x * 256 + threadIdx.x;
  if (i < 64 * 256) {
    int k = i / 256, n = i % 256;
    w1[i] = (_Float16)in_proj[dir * 256 * 64 + n * 64 + k];
  } else if (i < 64 * 256 + 128 * 48) {
    int j = i - 64 * 256;
    int k = j / 48, n = j % 48;
    w2[j] = (_Float16)(n < 36 ? x_proj[dir * 36 * 128 + n * 128 + k] : 0.0f);
  } else if (i < 64 * 256 + 128 * 48 + 128 * 64) {
    int j = i - 64 * 256 - 128 * 48;
    int k = j / 64, n = j % 64;
    w3[j] = (_Float16)out_proj[dir * 64 * 128 + n * 128 + k];
  }
}

// ---------------- gather + LayerNorm -> f16 tokens [TTOK x 64] ----------------
__global__ void k_ln(int dir, const float* __restrict__ x,
                     const float* __restrict__ ln_g, const float* __restrict__ ln_b,
                     _Float16* __restrict__ xn) {
  int wave = threadIdx.x >> 5;
  int lane = threadIdx.x & 31;
  int t = blockIdx.x * 8 + wave;
  int s = t / LSEQ, l = t % LSEQ;
  int c0 = lane * 2;
  float x0 = x[gidx(dir, s, l, c0)];
  float x1 = x[gidx(dir, s, l, c0 + 1)];
  float sum = x0 + x1;
  #pragma unroll
  for (int m = 16; m >= 1; m >>= 1) sum += __shfl_xor(sum, m, 32);
  float mean = sum * (1.0f / 64.0f);
  float d0 = x0 - mean, d1 = x1 - mean;
  float vs = d0 * d0 + d1 * d1;
  #pragma unroll
  for (int m = 16; m >= 1; m >>= 1) vs += __shfl_xor(vs, m, 32);
  float inv = rsqrtf(vs * (1.0f / 64.0f) + 1e-5f);
  xn[t * 64 + c0]     = (_Float16)(d0 * inv * ln_g[dir * 64 + c0]     + ln_b[dir * 64 + c0]);
  xn[t * 64 + c0 + 1] = (_Float16)(d1 * inv * ln_g[dir * 64 + c0 + 1] + ln_b[dir * 64 + c0 + 1]);
}

// ---------------- LDS-staged WMMA GEMM ----------------
// Kk: reduction dim, Nn: output cols, NTW: 16-col tiles per wave, MTB: 16-row tiles per block.
// B panel (Kk x Nn) and the block's A panel (MTB*16 x Kk) staged to LDS with
// global_load_async_to_lds_b128; A frags then come from ds_load_b128, B frags from 16-bit LDS reads.
// Per K-step: load A frag + ALL NTW B frags first, then issue NTW WMMAs back-to-back
// (single dscnt wait, independent accumulators cover the WMMA hazard windows).
template<int Kk, int Nn, int NTW, int MTB>
__global__ void k_gemm_t(const _Float16* __restrict__ A, const _Float16* __restrict__ B,
                         float* __restrict__ C) {
  __shared__ __align__(32) _Float16 Bs[Kk * Nn];
  __shared__ __align__(32) _Float16 As[MTB * 16 * Kk];
  const int WPM = (Nn / 16) / NTW;             // waves per m-tile
  int wave = threadIdx.x >> 5, lane = threadIdx.x & 31;

  coop_async_copy(B, lds_off_of(Bs), (Kk * Nn) / 8);
  coop_async_copy(A + (size_t)blockIdx.x * MTB * 16 * Kk, lds_off_of(As), (MTB * 16 * Kk) / 8);
  wait_async0();
  __syncthreads();

  int lmt = wave / WPM;                        // local m-tile
  int ntb = (wave % WPM) * NTW;                // first n-tile for this wave
  int g = lane >> 4, m = lane & 15;

  v8f acc[NTW];
  #pragma unroll
  for (int j = 0; j < NTW; j++) acc[j] = (v8f){};

  #pragma unroll
  for (int k0 = 0; k0 < Kk; k0 += 32) {
    const _Float16* ap = As + (lmt * 16 + m) * Kk + k0 + g * 8;
    v8h lo = *(const v8h*)ap;
    v8h hi = *(const v8h*)(ap + 16);
    v16h av;
    #pragma unroll
    for (int e = 0; e < 8; e++) { av[e] = lo[e]; av[8 + e] = hi[e]; }
    v16h bv[NTW];
    #pragma unroll
    for (int j = 0; j < NTW; j++)
      for (int e = 0; e < 16; e++)
        bv[j][e] = Bs[(k0 + 16 * (e >> 3) + 8 * g + (e & 7)) * Nn + (ntb + j) * 16 + m];
    #pragma unroll
    for (int j = 0; j < NTW; j++)
      acc[j] = __builtin_amdgcn_wmma_f32_16x16x32_f16(false, av, false, bv[j],
                                                      (short)0, acc[j], false, false);
  }
  int row0 = (blockIdx.x * MTB + lmt) * 16 + 8 * g;
  #pragma unroll
  for (int j = 0; j < NTW; j++) {
    int col = (ntb + j) * 16 + m;
    #pragma unroll
    for (int v = 0; v < 8; v++) C[(size_t)(row0 + v) * Nn + col] = acc[j][v];
  }
}

// ---------------- causal depthwise conv (K=4) + SiLU -> u (f16) ----------------
__global__ void k_conv(int dir, const float* __restrict__ xr,
                       const float* __restrict__ conv_w, const float* __restrict__ conv_b,
                       _Float16* __restrict__ u) {
  int i = blockIdx.x * 256 + threadIdx.x;
  int t = i >> 7, dd = i & 127;
  int l = t % LSEQ;
  float acc = conv_b[dir * DIx + dd];
  #pragma unroll
  for (int j = 0; j < KCONV; j++) {
    int ls = l - (KCONV - 1) + j;
    if (ls >= 0)
      acc += conv_w[(dir * DIx + dd) * KCONV + j] * xr[(t - (KCONV - 1) + j) * 256 + dd];
  }
  float sg = 1.0f / (1.0f + __expf(-acc));
  u[i] = (_Float16)(acc * sg);
}

// ---------------- delta = softplus(dlt @ dt_w.T + dt_b) ----------------
__global__ void k_delta(int dir, const float* __restrict__ xdbl,
                        const float* __restrict__ dt_w, const float* __restrict__ dt_b,
                        float* __restrict__ delta) {
  int i = blockIdx.x * 256 + threadIdx.x;
  int t = i >> 7, dd = i & 127;
  float acc = dt_b[dir * DIx + dd];
  #pragma unroll
  for (int r = 0; r < DTRANK; r++)
    acc += xdbl[t * 48 + r] * dt_w[(dir * DIx + dd) * DTRANK + r];
  delta[i] = (acc > 20.0f) ? acc : log1pf(__expf(acc));
}

// ---------------- selective scan: one wave32 per sequence ----------------
__global__ void k_scan(int dir, const float* __restrict__ xdbl,
                       const float* __restrict__ delta, const _Float16* __restrict__ u,
                       const float* __restrict__ xr, const float* __restrict__ A_log,
                       const float* __restrict__ D_ssm, _Float16* __restrict__ y) {
  int wave = threadIdx.x >> 5, lane = threadIdx.x & 31;
  int s = blockIdx.x * 8 + wave;
  float a[4][16], h[4][16], Dp[4];
  #pragma unroll
  for (int i = 0; i < 4; i++) {
    int dd = lane + 32 * i;
    Dp[i] = D_ssm[dir * DIx + dd];
    #pragma unroll
    for (int n = 0; n < 16; n++) {
      a[i][n] = -__expf(A_log[(dir * DIx + dd) * NST + n]);
      h[i][n] = 0.0f;
    }
  }
  for (int l = 0; l < LSEQ; l++) {
    int t = s * LSEQ + l;
    if (l + 1 < LSEQ) {   // overlap next step's HBM latency with this step's exp chain
      int tn = t + 1;
      __builtin_prefetch(&delta[tn * DIx + lane], 0, 0);
      __builtin_prefetch(&u[tn * DIx + lane], 0, 0);
      __builtin_prefetch(&xdbl[tn * 48 + (lane & 15)], 0, 0);
      __builtin_prefetch(&xr[tn * 256 + 128 + lane], 0, 0);
    }
    float Bv[16], Cv[16];
    #pragma unroll
    for (int n = 0; n < 16; n++) {
      Bv[n] = xdbl[t * 48 + 4 + n];
      Cv[n] = xdbl[t * 48 + 20 + n];
    }
    #pragma unroll
    for (int i = 0; i < 4; i++) {
      int dd = lane + 32 * i;
      float dt = delta[t * DIx + dd];
      float uu = (float)u[t * DIx + dd];
      float du = dt * uu;
      float acc = 0.0f;
      #pragma unroll
      for (int n = 0; n < 16; n++) {
        h[i][n] = __expf(dt * a[i][n]) * h[i][n] + du * Bv[n];
        acc += h[i][n] * Cv[n];
      }
      acc += uu * Dp[i];
      float r = xr[t * 256 + 128 + dd];
      acc *= r / (1.0f + __expf(-r));
      y[t * DIx + dd] = (_Float16)acc;
    }
  }
}

// ---------------- out_proj WMMA GEMM (LDS-staged) + weighted scatter ----------------
__global__ void k_gemm_out(int dir, const _Float16* __restrict__ A,
                           const _Float16* __restrict__ B,
                           const float* __restrict__ alpha, float* __restrict__ out) {
  const int Kk = 128, Nn = 64, NTW = 4, MTB = 8;
  __shared__ __align__(32) _Float16 Bs[Kk * Nn];
  __shared__ __align__(32) _Float16 As[MTB * 16 * Kk];
  int wave = threadIdx.x >> 5, lane = threadIdx.x & 31;

  coop_async_copy(B, lds_off_of(Bs), (Kk * Nn) / 8);
  coop_async_copy(A + (size_t)blockIdx.x * MTB * 16 * Kk, lds_off_of(As), (MTB * 16 * Kk) / 8);
  wait_async0();
  __syncthreads();

  int g = lane >> 4, m = lane & 15;
  v8f acc[NTW];
  #pragma unroll
  for (int j = 0; j < NTW; j++) acc[j] = (v8f){};

  #pragma unroll
  for (int k0 = 0; k0 < Kk; k0 += 32) {
    const _Float16* ap = As + (wave * 16 + m) * Kk + k0 + g * 8;
    v8h lo = *(const v8h*)ap;
    v8h hi = *(const v8h*)(ap + 16);
    v16h av;
    #pragma unroll
    for (int e = 0; e < 8; e++) { av[e] = lo[e]; av[8 + e] = hi[e]; }
    v16h bv[NTW];
    #pragma unroll
    for (int j = 0; j < NTW; j++)
      for (int e = 0; e < 16; e++)
        bv[j][e] = Bs[(k0 + 16 * (e >> 3) + 8 * g + (e & 7)) * Nn + j * 16 + m];
    #pragma unroll
    for (int j = 0; j < NTW; j++)
      acc[j] = __builtin_amdgcn_wmma_f32_16x16x32_f16(false, av, false, bv[j],
                                                      (short)0, acc[j], false, false);
  }
  float a0 = alpha[0], a1 = alpha[1], a2 = alpha[2];
  float mx = fmaxf(a0, fmaxf(a1, a2));
  float e0 = __expf(a0 - mx), e1 = __expf(a1 - mx), e2 = __expf(a2 - mx);
  float w = ((dir == 0) ? e0 : (dir == 1) ? e1 : e2) / (e0 + e1 + e2);
  int row0 = (blockIdx.x * MTB + wave) * 16 + 8 * g;
  #pragma unroll
  for (int j = 0; j < NTW; j++) {
    int ch = j * 16 + m;
    #pragma unroll
    for (int v = 0; v < 8; v++) {
      int t = row0 + v;
      int s = t / LSEQ, l = t % LSEQ;
      int addr = gidx(dir, s, l, ch);
      float val = w * acc[j][v];
      if (dir == 0) out[addr] = val;   // overwrite poisoned d_out
      else          out[addr] += val;  // accumulate (race-free per direction)
    }
  }
}

// ---------------- host launcher ----------------
static inline size_t alignup(size_t v) { return (v + 255) & ~(size_t)255; }

extern "C" void kernel_launch(void* const* d_in, const int* in_sizes, int n_in,
                              void* d_out, int out_size, void* d_ws, size_t ws_size,
                              hipStream_t stream) {
  (void)in_sizes; (void)n_in; (void)out_size; (void)ws_size;
  const float* x        = (const float*)d_in[0];
  const float* in_proj  = (const float*)d_in[1];
  const float* conv_w   = (const float*)d_in[2];
  const float* conv_b   = (const float*)d_in[3];
  const float* x_proj   = (const float*)d_in[4];
  const float* dt_w     = (const float*)d_in[5];
  const float* dt_b     = (const float*)d_in[6];
  const float* A_log    = (const float*)d_in[7];
  const float* D_ssm    = (const float*)d_in[8];
  const float* out_proj = (const float*)d_in[9];
  const float* ln_g     = (const float*)d_in[10];
  const float* ln_b     = (const float*)d_in[11];
  const float* alpha    = (const float*)d_in[12];
  float* out = (float*)d_out;
  char* ws = (char*)d_ws;

  size_t off = 0;
  _Float16* xn   = (_Float16*)(ws + off); off = alignup(off + (size_t)TTOK * 64 * 2);
  float*    xr   = (float*)   (ws + off); off = alignup(off + (size_t)TTOK * 256 * 4);
  _Float16* u    = (_Float16*)(ws + off); off = alignup(off + (size_t)TTOK * 128 * 2);
  float*    xdbl = (float*)   (ws + off); off = alignup(off + (size_t)TTOK * 48 * 4);
  float*    dlt  = (float*)   (ws + off); off = alignup(off + (size_t)TTOK * 128 * 4);
  _Float16* yb   = (_Float16*)(ws + off); off = alignup(off + (size_t)TTOK * 128 * 2);
  _Float16* w1   = (_Float16*)(ws + off); off = alignup(off + (size_t)64 * 256 * 2);
  _Float16* w2   = (_Float16*)(ws + off); off = alignup(off + (size_t)128 * 48 * 2);
  _Float16* w3   = (_Float16*)(ws + off); off = alignup(off + (size_t)128 * 64 * 2);

  for (int dir = 0; dir < 3; ++dir) {
    k_prep <<<120,    256, 0, stream>>>(dir, in_proj, x_proj, out_proj, w1, w2, w3);
    k_ln   <<<TTOK/8, 256, 0, stream>>>(dir, x, ln_g, ln_b, xn);
    // in_proj: M=TTOK, N=256, K=64 ; 2 m-tiles/block, 4 n-tiles/wave
    k_gemm_t<64, 256, 4, 2><<<TTOK/32,  256, 0, stream>>>(xn, w1, xr);
    k_conv <<<TTOK*128/256, 256, 0, stream>>>(dir, xr, conv_w, conv_b, u);
    // x_proj: M=TTOK, N=48, K=128 ; 8 m-tiles/block, 3 n-tiles/wave
    k_gemm_t<128, 48, 3, 8><<<TTOK/128, 256, 0, stream>>>(u, w2, xdbl);
    k_delta<<<TTOK*128/256, 256, 0, stream>>>(dir, xdbl, dt_w, dt_b, dlt);
    k_scan <<<SEQS/8, 256, 0, stream>>>(dir, xdbl, dlt, u, xr, A_log, D_ssm, yb);
    // out_proj: M=TTOK, N=64, K=128 ; 8 m-tiles/block, 4 n-tiles/wave
    k_gemm_out<<<TTOK/128, 256, 0, stream>>>(dir, yb, w3, alpha, out);
  }
}
